// SwinBlock_7026566496908
// MI455X (gfx1250) — compile-verified, matrix operand order checked by reading, not run on hardware
//
#include <hip/hip_runtime.h>
#include <cstdint>
#include <cstddef>

typedef __attribute__((ext_vector_type(16))) __bf16 v16bf;
typedef __attribute__((ext_vector_type(8)))  float  v8f;

static __device__ __forceinline__ v8f wmma_bf16(v16bf a, v16bf b, v8f c) {
  // D = A(16x32 bf16) * B(32x16 bf16) + C(16x16 f32)
  return __builtin_amdgcn_wmma_f32_16x16x32_bf16(false, a, false, b, (short)0, c, false, false);
}

// A-fragment loader (16-bit, 16x32, wave32 layout, ISA 7.12.2):
//   lanes 0-15: row = lane,    K pairs {0..7}  in v0-3, {16..23} in v4-7
//   lanes16-31: row = lane-16, K pairs {8..15} in v0-3, {24..31} in v4-7
// Also serves as B-fragment loader for B supplied as B^T[N][K] (lane = column).
static __device__ __forceinline__ v16bf load_frag(const __bf16* p, int ld) {
  const int lane = threadIdx.x & 31;
  const int half = lane >> 4;
  const int r    = lane & 15;
  union { uint32_t u[8]; v16bf v; } f;
  const uint32_t* q = reinterpret_cast<const uint32_t*>(p + (size_t)r * ld);
#pragma unroll
  for (int j = 0; j < 8; ++j) {
    const int kb = (j < 4) ? (2 * j + 8 * half) : (16 + 2 * (j - 4) + 8 * half);
    f.u[j] = q[kb >> 1];
  }
  return f.v;
}

// Same, but rows >= rmax are zero-filled (padding 49 -> 64).
static __device__ __forceinline__ v16bf load_frag_pred(const __bf16* p, int ld, int row0, int rmax) {
  const int lane = threadIdx.x & 31;
  const int half = lane >> 4;
  const int r    = row0 + (lane & 15);
  union { uint32_t u[8]; v16bf v; } f;
  if (r < rmax) {
    const uint32_t* q = reinterpret_cast<const uint32_t*>(p + (size_t)r * ld);
#pragma unroll
    for (int j = 0; j < 8; ++j) {
      const int kb = (j < 4) ? (2 * j + 8 * half) : (16 + 2 * (j - 4) + 8 * half);
      f.u[j] = q[kb >> 1];
    }
  } else {
#pragma unroll
    for (int j = 0; j < 8; ++j) f.u[j] = 0u;
  }
  return f.v;
}

// ---------------- Weight convert: dst[n*K+k] = (bf16) src[k*N+n]  (B^T layout) ----------
__global__ __launch_bounds__(256) void wconv_kernel(const float* __restrict__ src,
                                                    __bf16* __restrict__ dst, int K, int N) {
  const size_t i = (size_t)blockIdx.x * 256 + threadIdx.x;
  if (i >= (size_t)K * N) return;
  const int n = (int)(i / (size_t)K);
  const int k = (int)(i - (size_t)n * K);
  dst[i] = (__bf16)src[(size_t)k * N + n];
}

// ---------------- LN1 fused with roll(-3,-3) + 7x7 window partition (gather) ------------
__global__ __launch_bounds__(256) void ln1_window_kernel(const float* __restrict__ x,
                                                         const float* __restrict__ g,
                                                         const float* __restrict__ b,
                                                         __bf16* __restrict__ xw) {
  const int t = blockIdx.x;         // windowed token index
  const int ch = threadIdx.x;       // channel
  const int win = t / 49, n = t - win * 49;
  const int bimg = win >> 10, wl = win & 1023;
  const int wi = wl >> 5, wj = wl & 31;
  const int a = n / 7, bc = n - a * 7;
  int rr = wi * 7 + a + 3;  if (rr >= 224) rr -= 224;   // shifted coord -> source coord
  int cc = wj * 7 + bc + 3; if (cc >= 224) cc -= 224;
  const size_t src = ((size_t)bimg * 50176 + (size_t)rr * 224 + cc) * 256 + ch;
  const float v = x[src];
  float s = v, s2 = v * v;
#pragma unroll
  for (int off = 1; off < 32; off <<= 1) { s += __shfl_xor(s, off, 32); s2 += __shfl_xor(s2, off, 32); }
  __shared__ float sh[2][8];
  const int wave = threadIdx.x >> 5, lane = threadIdx.x & 31;
  if (lane == 0) { sh[0][wave] = s; sh[1][wave] = s2; }
  __syncthreads();
  float ts = 0.f, ts2 = 0.f;
#pragma unroll
  for (int i = 0; i < 8; ++i) { ts += sh[0][i]; ts2 += sh[1][i]; }
  const float mu  = ts * (1.f / 256.f);
  const float var = ts2 * (1.f / 256.f) - mu * mu;
  const float y = (v - mu) * rsqrtf(var + 1e-5f) * g[ch] + b[ch];
  xw[(size_t)t * 256 + ch] = (__bf16)y;
}

// ---------------- LN2 (token-major) -----------------------------------------------------
__global__ __launch_bounds__(256) void ln2_kernel(const float* __restrict__ xin,
                                                  const float* __restrict__ g,
                                                  const float* __restrict__ b,
                                                  __bf16* __restrict__ h1) {
  const int t = blockIdx.x;
  const int ch = threadIdx.x;
  const float v = xin[(size_t)t * 256 + ch];
  float s = v, s2 = v * v;
#pragma unroll
  for (int off = 1; off < 32; off <<= 1) { s += __shfl_xor(s, off, 32); s2 += __shfl_xor(s2, off, 32); }
  __shared__ float sh[2][8];
  const int wave = threadIdx.x >> 5, lane = threadIdx.x & 31;
  if (lane == 0) { sh[0][wave] = s; sh[1][wave] = s2; }
  __syncthreads();
  float ts = 0.f, ts2 = 0.f;
#pragma unroll
  for (int i = 0; i < 8; ++i) { ts += sh[0][i]; ts2 += sh[1][i]; }
  const float mu  = ts * (1.f / 256.f);
  const float var = ts2 * (1.f / 256.f) - mu * mu;
  const float y = (v - mu) * rsqrtf(var + 1e-5f) * g[ch] + b[ch];
  h1[(size_t)t * 256 + ch] = (__bf16)y;
}

// ---------------- Generic WMMA GEMM: C[M,N] = A[M,K] * B (as B^T[N][K]) + bias ----------
// Per-wave tile: 32(M) x 64(N) = 2x4 WMMA tiles; 8 WMMA per 12 b128 loads per k-step.
// mode 0: qkv  -> scale q-columns (<256) by HEAD_DIM^-0.5, bf16 out
// mode 1: proj -> window-reverse + roll(+3,+3) scatter, + residual x, f32 out (d_out)
// mode 2: fc1  -> exact GELU, bf16 out
// mode 3: fc2  -> accumulate into f32 out (d_out second residual)
__global__ __launch_bounds__(256) void gemm_kernel(const __bf16* __restrict__ A,
                                                   const __bf16* __restrict__ Bt,
                                                   const float* __restrict__ bias,
                                                   int Ncols, int K, int mode,
                                                   __bf16* __restrict__ out_bf,
                                                   float* __restrict__ out_f,
                                                   const float* __restrict__ resid) {
  const int wave = threadIdx.x >> 5;
  const int lane = threadIdx.x & 31;
  const size_t row0 = (size_t)blockIdx.x * 256 + (size_t)wave * 32;
  const int col0 = blockIdx.y * 64;
  const __bf16* Ap0 = A + row0 * (size_t)K;
  const __bf16* Ap1 = Ap0 + (size_t)16 * K;
  const __bf16* Bp[4];
#pragma unroll
  for (int nt = 0; nt < 4; ++nt) Bp[nt] = Bt + (size_t)(col0 + nt * 16) * K;
  const v8f vzero = {0.f, 0.f, 0.f, 0.f, 0.f, 0.f, 0.f, 0.f};
  v8f acc[2][4];
#pragma unroll
  for (int mt = 0; mt < 2; ++mt)
#pragma unroll
    for (int nt = 0; nt < 4; ++nt) acc[mt][nt] = vzero;
  for (int k0 = 0; k0 < K; k0 += 32) {
    __builtin_prefetch(Ap0 + k0 + 64, 0, 0);
    __builtin_prefetch(Ap1 + k0 + 64, 0, 0);
    v16bf af0 = load_frag(Ap0 + k0, K);
    v16bf af1 = load_frag(Ap1 + k0, K);
#pragma unroll
    for (int nt = 0; nt < 4; ++nt) {
      v16bf bfrag = load_frag(Bp[nt] + k0, K);
      acc[0][nt] = wmma_bf16(af0, bfrag, acc[0][nt]);
      acc[1][nt] = wmma_bf16(af1, bfrag, acc[1][nt]);
    }
  }
  const int halfrow = (lane >> 4) * 8;
  const int lcol = lane & 15;
#pragma unroll
  for (int mt = 0; mt < 2; ++mt) {
#pragma unroll
    for (int nt = 0; nt < 4; ++nt) {
      const int col = col0 + nt * 16 + lcol;
      const float bv = bias[col];
#pragma unroll
      for (int r = 0; r < 8; ++r) {
        const size_t row = row0 + (size_t)(mt * 16 + r + halfrow);
        float v = acc[mt][nt][r] + bv;
        if (mode == 0) {
          if (col < 256) v *= 0.17677669529663687f;   // HEAD_DIM^-0.5 folded into q
          out_bf[row * Ncols + col] = (__bf16)v;
        } else if (mode == 2) {
          v = 0.5f * v * (1.0f + erff(v * 0.7071067811865475f));
          out_bf[row * Ncols + col] = (__bf16)v;
        } else if (mode == 3) {
          out_f[row * Ncols + col] += v;
        } else {  // mode 1: proj + window_reverse + unshift + residual
          const int t = (int)row;
          const int win = t / 49, n = t - win * 49;
          const int bimg = win >> 10, wl = win & 1023;
          const int wi = wl >> 5, wj = wl & 31;
          const int a = n / 7, bc = n - a * 7;
          int rr = wi * 7 + a + 3;  if (rr >= 224) rr -= 224;
          int cc = wj * 7 + bc + 3; if (cc >= 224) cc -= 224;
          const size_t dst = ((size_t)bimg * 50176 + (size_t)rr * 224 + cc) * 256 + col;
          out_f[dst] = resid[dst] + v;
        }
      }
    }
  }
}

// ---------------- Windowed attention: one wave = one (window, head) ---------------------
__global__ __launch_bounds__(128) void attn_kernel(const __bf16* __restrict__ qkv,
                                                   __bf16* __restrict__ attn_out) {
  __shared__ __bf16 lds_p[4][64 * 64];    // P (softmax probs), per wave
  __shared__ __bf16 lds_vt[4][32 * 64];   // V^T, per wave
  const int w = blockIdx.x;
  const int wave = threadIdx.x >> 5;
  const int lane = threadIdx.x & 31;
  const int head = blockIdx.y * 4 + wave;
  const __bf16* base = qkv + (size_t)w * 49 * 768;
  const __bf16* qp = base + head * 32;          // q: cols [0,256)
  const __bf16* kp = base + 256 + head * 32;    // k: cols [256,512)
  const __bf16* vp = base + 512 + head * 32;    // v: cols [512,768)

  // S = q @ k^T  (q pre-scaled in qkv epilogue). 64x64 padded, K=32 in one step.
  v16bf qf[4], kf[4];
#pragma unroll
  for (int mt = 0; mt < 4; ++mt) qf[mt] = load_frag_pred(qp, 768, mt * 16, 49);
#pragma unroll
  for (int nt = 0; nt < 4; ++nt) kf[nt] = load_frag_pred(kp, 768, nt * 16, 49);
  const v8f vzero = {0.f, 0.f, 0.f, 0.f, 0.f, 0.f, 0.f, 0.f};
  v8f S[4][4];
#pragma unroll
  for (int mt = 0; mt < 4; ++mt)
#pragma unroll
    for (int nt = 0; nt < 4; ++nt) S[mt][nt] = wmma_bf16(qf[mt], kf[nt], vzero);

  // Stage V^T into LDS: lds_vt[d][m] = v[m][d]   (makes P@V B-fragments contiguous)
#pragma unroll
  for (int m = lane; m < 64; m += 32) {
    const bool valid = m < 49;
    const uint32_t* vr = reinterpret_cast<const uint32_t*>(vp + (size_t)m * 768);
#pragma unroll
    for (int dd = 0; dd < 16; ++dd) {
      union { uint32_t uu; __bf16 h[2]; } cv;
      cv.uu = valid ? vr[dd] : 0u;
      lds_vt[wave][(2 * dd + 0) * 64 + m] = cv.h[0];
      lds_vt[wave][(2 * dd + 1) * 64 + m] = cv.h[1];
    }
  }

  // Shift-mask (only window (31,31) has the 8-valued 3x3 corner region) + softmax.
  const bool special = ((w & 1023) == 1023);
#pragma unroll
  for (int mt = 0; mt < 4; ++mt) {
#pragma unroll
    for (int r = 0; r < 8; ++r) {
      const int row = mt * 16 + r + ((lane >> 4) * 8);
      int rreg = 0;
      if (special && row < 49) rreg = ((row / 7) >= 4 && (row % 7) >= 4) ? 1 : 0;
      float vals[4];
      float mx = -3.0e38f;
#pragma unroll
      for (int nt = 0; nt < 4; ++nt) {
        const int col = nt * 16 + (lane & 15);
        float s = S[mt][nt][r];
        if (col >= 49) {
          s = -3.0e38f;                       // key padding
        } else if (special) {
          const int creg = ((col / 7) >= 4 && (col % 7) >= 4) ? 1 : 0;
          if (creg != rreg) s -= 100.0f;      // attn_mask
        }
        vals[nt] = s;
        mx = fmaxf(mx, s);
      }
#pragma unroll
      for (int off = 1; off < 16; off <<= 1) mx = fmaxf(mx, __shfl_xor(mx, off, 32));
      float sum = 0.f;
#pragma unroll
      for (int nt = 0; nt < 4; ++nt) { const float p = __expf(vals[nt] - mx); vals[nt] = p; sum += p; }
#pragma unroll
      for (int off = 1; off < 16; off <<= 1) sum += __shfl_xor(sum, off, 32);
      const float inv = 1.0f / sum;
#pragma unroll
      for (int nt = 0; nt < 4; ++nt) {
        const int col = nt * 16 + (lane & 15);
        lds_p[wave][row * 64 + col] = (__bf16)(vals[nt] * inv);
      }
    }
  }
  __syncthreads();

  // O = P @ V : P 64x64 (A-frags from LDS), V 64x32 (B-frags from V^T in LDS)
  v16bf vf[2][2];
#pragma unroll
  for (int kt = 0; kt < 2; ++kt)
#pragma unroll
    for (int nt = 0; nt < 2; ++nt)
      vf[kt][nt] = load_frag(&lds_vt[wave][(nt * 16) * 64 + kt * 32], 64);
  v8f O[4][2];
#pragma unroll
  for (int mt = 0; mt < 4; ++mt) {
    O[mt][0] = vzero; O[mt][1] = vzero;
#pragma unroll
    for (int kt = 0; kt < 2; ++kt) {
      v16bf pf = load_frag(&lds_p[wave][(mt * 16) * 64 + kt * 32], 64);
#pragma unroll
      for (int nt = 0; nt < 2; ++nt) O[mt][nt] = wmma_bf16(pf, vf[kt][nt], O[mt][nt]);
    }
  }
  // Store to attn_out[t][head*32 + d]
#pragma unroll
  for (int mt = 0; mt < 4; ++mt)
#pragma unroll
    for (int nt = 0; nt < 2; ++nt)
#pragma unroll
      for (int r = 0; r < 8; ++r) {
        const int row = mt * 16 + r + ((lane >> 4) * 8);
        const int col = nt * 16 + (lane & 15);
        if (row < 49)
          attn_out[((size_t)w * 49 + row) * 256 + head * 32 + col] = (__bf16)O[mt][nt][r];
      }
}

// ---------------------------------------------------------------------------------------
extern "C" void kernel_launch(void* const* d_in, const int* in_sizes, int n_in,
                              void* d_out, int out_size, void* d_ws, size_t ws_size,
                              hipStream_t stream) {
  const float* x      = (const float*)d_in[0];
  const float* qkv_w  = (const float*)d_in[3];
  const float* qkv_b  = (const float*)d_in[4];
  const float* proj_w = (const float*)d_in[5];
  const float* proj_b = (const float*)d_in[6];
  const float* n1w    = (const float*)d_in[7];
  const float* n1b    = (const float*)d_in[8];
  const float* n2w    = (const float*)d_in[9];
  const float* n2b    = (const float*)d_in[10];
  const float* fc1_w  = (const float*)d_in[11];
  const float* fc1_b  = (const float*)d_in[12];
  const float* fc2_w  = (const float*)d_in[13];
  const float* fc2_b  = (const float*)d_in[14];
  float* out = (float*)d_out;
  (void)in_sizes; (void)n_in; (void)out_size; (void)ws_size;

  const size_t M = 200704;  // 4 * 224 * 224 tokens (= 4096 windows * 49)
  char* ws = (char*)d_ws;
  size_t off = 0;
  auto take = [&](size_t bytes) { char* p = ws + off; off = (off + bytes + 255) & ~(size_t)255; return p; };
  __bf16* xw   = (__bf16*)take(M * 256 * 2);    // LN1 + windowed, bf16
  __bf16* qkv  = (__bf16*)take(M * 768 * 2);    // q|k|v, q pre-scaled
  __bf16* attn = (__bf16*)take(M * 256 * 2);    // attention output
  __bf16* h1   = (__bf16*)take(M * 256 * 2);    // LN2 output
  __bf16* h2   = (__bf16*)take(M * 1024 * 2);   // GELU(fc1)
  __bf16* wqT  = (__bf16*)take((size_t)768 * 256 * 2);
  __bf16* wpT  = (__bf16*)take((size_t)256 * 256 * 2);
  __bf16* w1T  = (__bf16*)take((size_t)1024 * 256 * 2);
  __bf16* w2T  = (__bf16*)take((size_t)256 * 1024 * 2);

  // Weight convert + transpose to B^T[N][K] bf16 (tiny; stays in L2)
  wconv_kernel<<<(768 * 256 + 255) / 256, 256, 0, stream>>>(qkv_w, wqT, 256, 768);
  wconv_kernel<<<(256 * 256 + 255) / 256, 256, 0, stream>>>(proj_w, wpT, 256, 256);
  wconv_kernel<<<(1024 * 256 + 255) / 256, 256, 0, stream>>>(fc1_w, w1T, 256, 1024);
  wconv_kernel<<<(1024 * 256 + 255) / 256, 256, 0, stream>>>(fc2_w, w2T, 1024, 256);

  // 1) LN1 + shift + window partition
  ln1_window_kernel<<<(int)M, 256, 0, stream>>>(x, n1w, n1b, xw);
  // 2) QKV GEMM (q scaled)
  gemm_kernel<<<dim3(784, 12), 256, 0, stream>>>(xw, wqT, qkv_b, 768, 256, 0, qkv, nullptr, nullptr);
  // 3) Windowed attention (4096 windows x 8 heads; one wave per head)
  attn_kernel<<<dim3(4096, 2), 128, 0, stream>>>(qkv, attn);
  // 4) proj GEMM + window reverse + unshift + residual -> d_out (f32)
  gemm_kernel<<<dim3(784, 4), 256, 0, stream>>>(attn, wpT, proj_b, 256, 256, 1, nullptr, out, x);
  // 5) LN2
  ln2_kernel<<<(int)M, 256, 0, stream>>>(out, n2w, n2b, h1);
  // 6) fc1 + GELU
  gemm_kernel<<<dim3(784, 16), 256, 0, stream>>>(h1, w1T, fc1_b, 1024, 256, 2, h2, nullptr, nullptr);
  // 7) fc2 + second residual accumulate into d_out
  gemm_kernel<<<dim3(784, 4), 256, 0, stream>>>(h2, w2T, fc2_b, 256, 1024, 3, nullptr, out, nullptr);
}
